// CrossAttention_37572373906047
// MI455X (gfx1250) — matrix operand order repaired and checked by reading, not compiled, not run on hardware
//
#include <hip/hip_runtime.h>

#define B_    2
#define NSEQ  2048
#define DIM_  1024
#define H_    16
#define DH_   64
#define INNER_ 1024
#define SCALE_ 0.125f

typedef __attribute__((ext_vector_type(16))) __bf16 bf16x16;
typedef __attribute__((ext_vector_type(8)))  float  f32x8;
typedef __attribute__((ext_vector_type(4)))  float  f32x4;
typedef unsigned short u16;
typedef unsigned int   u32;
typedef __attribute__((ext_vector_type(4))) u32 u32x4;
typedef __attribute__((ext_vector_type(2))) u32 u32x2;
typedef __attribute__((ext_vector_type(4))) int v4i;

union FragAB { bf16x16 v; u32 u[8]; };

#if __has_builtin(__builtin_amdgcn_global_load_async_to_lds_b128) && __has_builtin(__builtin_amdgcn_s_wait_asynccnt)
#define ASYNC_LDS 1
typedef __attribute__((address_space(1))) v4i* gptr1;
typedef __attribute__((address_space(3))) v4i* lptr3;
#else
#define ASYNC_LDS 0
#endif

__device__ inline void async_g2l_b128(const void* g, void* l) {
#if ASYNC_LDS
    __builtin_amdgcn_global_load_async_to_lds_b128(
        (gptr1)(unsigned long long)g,
        (lptr3)(unsigned)(unsigned long long)l, 0, 0);
#else
    (void)g; (void)l;
#endif
}
__device__ inline void wait_async0() {
#if ASYNC_LDS
    __builtin_amdgcn_s_wait_asynccnt(0);
#endif
}

__device__ inline u16 f2bf(float x) {
    u32 u = __float_as_uint(x);
    return (u16)((u + 0x7fffu + ((u >> 16) & 1u)) >> 16);
}
__device__ inline u32 f2bf2(float a, float b) { return (u32)f2bf(a) | ((u32)f2bf(b) << 16); }
__device__ inline u32 ldpair(const u16* p) { u32 v; __builtin_memcpy(&v, p, 4); return v; }

// Fragment from two 16B-aligned 16-byte chunks (global_load_b128 / ds_load_b128)
__device__ inline bf16x16 frag2(const u16* p0, const u16* p1) {
    FragAB fr;
    *(u32x4*)&fr.u[0] = *(const u32x4*)p0;
    *(u32x4*)&fr.u[4] = *(const u32x4*)p1;
    return fr.v;
}
// A-matrix 16x32 bf16 fragment, K contiguous per lane half (k = 16*half + s), stride in u16.
__device__ inline bf16x16 build_a_c(int lane, const u16* base, int stride, int k0) {
    int half = (lane >> 4) & 1, m = lane & 15;
    const u16* p = base + (size_t)m * stride + k0 + 16 * half;
    return frag2(p, p + 8);
}
// B-matrix 32x16 bf16 fragment, K contiguous per column row.
__device__ inline bf16x16 build_b_c(int lane, const u16* base, int stride, int k0) {
    int n = lane & 15, kb = (lane >> 4) * 16;
    const u16* p = base + (size_t)n * stride + k0 + kb;
    return frag2(p, p + 8);
}
// generic lambda builders (used for talking-heads fragments / LDS prob planes)
template <typename F>
__device__ inline bf16x16 build_a(int lane, F f) {
    FragAB fr; int half = (lane >> 4) & 1, m = lane & 15;
#pragma unroll
    for (int r = 0; r < 8; ++r) {
        int kp = 2 * r + 16 * half;
        fr.u[r] = f(m, kp);
    }
    return fr.v;
}
template <typename F>
__device__ inline bf16x16 build_b(int lane, F f) {
    FragAB fr; int n = lane & 15, kb = (lane >> 4) * 16;
#pragma unroll
    for (int r = 0; r < 8; ++r) fr.u[r] = f(kb + 2 * r, n);
    return fr.v;
}
__device__ inline f32x8 wmma_bf16(bf16x16 a, bf16x16 b, f32x8 c) {
    return __builtin_amdgcn_wmma_f32_16x16x32_bf16(false, a, false, b, (short)0, c, false, false);
}

// ---------------- elementwise / LN kernels ----------------
__global__ void __launch_bounds__(256) k_zero4(f32x4* p, int n4) {
    int i = blockIdx.x * 256 + threadIdx.x;
    if (i < n4) { f32x4 z = {0.f, 0.f, 0.f, 0.f}; p[i] = z; }
}
__global__ void __launch_bounds__(256) k_f32bf(const float* __restrict__ s, u16* __restrict__ d, int n) {
    int base = (blockIdx.x * 256 + threadIdx.x) * 4;
    if (base < n) {
        f32x4 v = *(const f32x4*)(s + base);
        u32x2 o = {f2bf2(v[0], v[1]), f2bf2(v[2], v[3])};
        *(u32x2*)(d + base) = o;
    }
}
__global__ void __launch_bounds__(256) k_layernorm(const float* __restrict__ x,
                                                   const float* __restrict__ g,
                                                   const float* __restrict__ bta,
                                                   u16* __restrict__ out) {
    int row = blockIdx.x, t = threadIdx.x;
    const float* xr = x + (size_t)row * DIM_;
    f32x4 v = *(const f32x4*)(xr + t * 4);
    float lsum = v[0] + v[1] + v[2] + v[3];
    float lsq = v[0] * v[0] + v[1] * v[1] + v[2] * v[2] + v[3] * v[3];
    __shared__ float s1[256], s2[256];
    s1[t] = lsum; s2[t] = lsq; __syncthreads();
    for (int o = 128; o > 0; o >>= 1) {
        if (t < o) { s1[t] += s1[t + o]; s2[t] += s2[t + o]; }
        __syncthreads();
    }
    float mu = s1[0] * (1.f / DIM_);
    float var = s2[0] * (1.f / DIM_) - mu * mu;
    float rstd = rsqrtf(var + 1e-5f);
    f32x4 gv = *(const f32x4*)(g + t * 4);
    f32x4 bv = *(const f32x4*)(bta + t * 4);
    float y0 = (v[0] - mu) * rstd * gv[0] + bv[0];
    float y1 = (v[1] - mu) * rstd * gv[1] + bv[1];
    float y2 = (v[2] - mu) * rstd * gv[2] + bv[2];
    float y3 = (v[3] - mu) * rstd * gv[3] + bv[3];
    u32x2 o = {f2bf2(y0, y1), f2bf2(y2, y3)};
    *(u32x2*)(out + (size_t)row * DIM_ + t * 4) = o;
}

// ---------------- generic bf16 GEMM: C[M,N] = A[M,K] * W[K,N] ----------------
// Double-buffered LDS tiles; A staged via async-to-LDS b128 when available.
// mode 0: bf16 out in head-split layout [b,h,n,d]   mode 1: f32 out row-major + bias
__global__ void __launch_bounds__(128) k_gemm(const u16* __restrict__ A, const u16* __restrict__ W,
                                              int K, int Ncols, int mode,
                                              const float* __restrict__ bias,
                                              float* __restrict__ outF, u16* __restrict__ outH) {
    __shared__ u32 As32[2][1024];                  // A tile 64x32 bf16, double-buffered
    __shared__ __align__(16) u16 Ws[2][64 * 40];   // W tile transposed [n][k], 80B row stride
    int t = threadIdx.x, lane = t & 31, w = t >> 5;
    int i0 = blockIdx.x * 64, n0 = blockIdx.y * 64;
    f32x8 acc[4] = {};

    auto stage = [&](int k0, int p) {
        int rr = t >> 1, hh = t & 1;
        const u16* ga = A + (size_t)(i0 + rr) * K + k0 + hh * 16;
        u16* la = (u16*)As32[p] + rr * 32 + hh * 16;
#if ASYNC_LDS
        async_g2l_b128(ga, la);
        async_g2l_b128(ga + 8, la + 8);
#else
        *(u32x4*)la = *(const u32x4*)ga;
        *(u32x4*)(la + 8) = *(const u32x4*)(ga + 8);
#endif
        int kk = t >> 2, c4 = t & 3;
        const u16* gw = W + (size_t)(k0 + kk) * Ncols + n0 + c4 * 16;
        u32x4 w0 = *(const u32x4*)gw;
        u32x4 w1 = *(const u32x4*)(gw + 8);
        const u16* e0 = (const u16*)&w0;
        const u16* e1 = (const u16*)&w1;
#pragma unroll
        for (int e = 0; e < 8; ++e) {
            Ws[p][(c4 * 16 + e) * 40 + kk] = e0[e];
            Ws[p][(c4 * 16 + 8 + e) * 40 + kk] = e1[e];
        }
        if (k0 + 64 < K) __builtin_prefetch(A + (size_t)(i0 + rr) * K + k0 + 64, 0, 3);
    };

    stage(0, 0);
    int p = 0;
    for (int k0 = 0; k0 < K; k0 += 32) {
        wait_async0();
        __syncthreads();
        if (k0 + 32 < K) stage(k0 + 32, p ^ 1);
        const u16* Asl = (const u16*)As32[p];
        bf16x16 af = build_a_c(lane, Asl + (w * 16) * 32, 32, 0);
        bf16x16 bfr[4];
#pragma unroll
        for (int s = 0; s < 4; ++s) bfr[s] = build_b_c(lane, &Ws[p][(s * 16) * 40], 40, 0);
#pragma unroll
        for (int s = 0; s < 4; ++s) acc[s] = wmma_bf16(af, bfr[s], acc[s]);
        p ^= 1;
    }
    int half = (lane >> 4) & 1, nl = lane & 15;
#pragma unroll
    for (int s = 0; s < 4; ++s)
#pragma unroll
        for (int r = 0; r < 8; ++r) {
            int gr = i0 + w * 16 + r + 8 * half;
            int gc = n0 + s * 16 + nl;
            if (mode == 0) {
                int bb = gr >> 11, nn = gr & 2047, h = gc >> 6, d = gc & 63;
                outH[(((size_t)(bb * H_ + h)) * NSEQ + nn) * DH_ + d] = f2bf(acc[s][r]);
            } else {
                outF[(size_t)gr * Ncols + gc] = acc[s][r] + bias[gc];
            }
        }
}

// ---------------- pass 1: exp-sum over rows (j) and cols (i) ----------------
__global__ void __launch_bounds__(128) k_attn_sums(const u16* __restrict__ qk, const u16* __restrict__ cqk,
                                                   float* __restrict__ rowsum, float* __restrict__ colsum) {
    int bh = blockIdx.x, i0 = blockIdx.y * 64;
    int t = threadIdx.x, lane = t & 31, w = t >> 5, half = (lane >> 4) & 1, nl = lane & 15;
    __shared__ float rs[64];
    if (t < 64) rs[t] = 0.f;
    __syncthreads();
    const u16* qb = qk + (size_t)bh * NSEQ * DH_;
    const u16* cb = cqk + (size_t)bh * NSEQ * DH_;
    float rowacc[8] = {0.f, 0.f, 0.f, 0.f, 0.f, 0.f, 0.f, 0.f};
    for (int j0 = 0; j0 < NSEQ; j0 += 64) {
        f32x8 acc[4] = {};
#pragma unroll
        for (int k0 = 0; k0 < DH_; k0 += 32) {
            bf16x16 af = build_a_c(lane, qb + (size_t)(i0 + w * 16) * DH_, DH_, k0);
            bf16x16 bfr[4];
#pragma unroll
            for (int s = 0; s < 4; ++s) bfr[s] = build_b_c(lane, cb + (size_t)(j0 + s * 16) * DH_, DH_, k0);
#pragma unroll
            for (int s = 0; s < 4; ++s) acc[s] = wmma_bf16(af, bfr[s], acc[s]);
        }
#pragma unroll
        for (int s = 0; s < 4; ++s) {
            float cp = 0.f;
#pragma unroll
            for (int r = 0; r < 8; ++r) {
                float e = __expf(acc[s][r] * SCALE_);
                rowacc[r] += e; cp += e;
            }
            atomicAdd(&colsum[(size_t)bh * NSEQ + j0 + s * 16 + nl], cp);
        }
    }
#pragma unroll
    for (int r = 0; r < 8; ++r) atomicAdd(&rs[w * 16 + r + 8 * half], rowacc[r]);
    __syncthreads();
    if (t < 64) rowsum[(size_t)bh * NSEQ + i0 + t] = rs[t];
}

// ---------------- pass 2: normalize, talking-heads mix (WMMA), P@V / Pc^T@V ----------------
// ~297KB of the WGP's 320KB LDS: 16 head prob planes + 16 mixed planes per 64x64 tile.
__global__ void __launch_bounds__(128) k_attn_apply(const u16* __restrict__ qk, const u16* __restrict__ cqk,
                                                    const u16* __restrict__ vv, const u16* __restrict__ cv,
                                                    const float* __restrict__ rowsum,
                                                    const float* __restrict__ colsum,
                                                    const float* __restrict__ thw,
                                                    const float* __restrict__ cthw,
                                                    float* __restrict__ out_acc, float* __restrict__ ctx_acc) {
    extern __shared__ char smem[];
    u16* ldsP = (u16*)smem;              // [4096 elems][18 pad] heads contiguous: 147456 B
    u16* ldsM = ldsP + 4096 * 18;        // 16 mixed planes [g][4112 pad]:        131584 B
    u16* Vt = ldsM + 16 * 4112;          // V tile transposed [d][72 pad]:          9216 B
    float* rinv = (float*)(Vt + 64 * 72);// [h][64]                                 4096 B
    float* cinv = rinv + 1024;           //                                         4096 B

    int b = blockIdx.y, tile = blockIdx.x;
    int i0 = (tile & 31) * 64, j0 = (tile >> 5) * 64;
    int t = threadIdx.x, lane = t & 31, w = t >> 5, half = (lane >> 4) & 1, nl = lane & 15;

    for (int q = t; q < 1024; q += 128) {
        int h = q >> 6, ii = q & 63;
        rinv[q] = 1.f / rowsum[((size_t)(b * H_ + h)) * NSEQ + i0 + ii];
        cinv[q] = 1.f / colsum[((size_t)(b * H_ + h)) * NSEQ + j0 + ii];
    }
    // talking-heads A fragments: 16(g) x 32(h, zero-padded K=16..31)
    bf16x16 thfrag = build_a(lane, [&](int m, int kp) {
        return kp < 16 ? f2bf2(thw[m * 16 + kp], thw[m * 16 + kp + 1]) : 0u; });
    bf16x16 cthfrag = build_a(lane, [&](int m, int kp) {
        return kp < 16 ? f2bf2(cthw[m * 16 + kp], cthw[m * 16 + kp + 1]) : 0u; });
    __syncthreads();

    for (int side = 0; side < 2; ++side) {
        // Phase A: recompute E per head, normalize, stage to LDS (heads contiguous per element)
        for (int h = 0; h < H_; ++h) {
            const u16* qb = qk + ((size_t)(b * H_ + h)) * NSEQ * DH_;
            const u16* cb = cqk + ((size_t)(b * H_ + h)) * NSEQ * DH_;
            f32x8 acc[4] = {};
#pragma unroll
            for (int k0 = 0; k0 < DH_; k0 += 32) {
                bf16x16 af = build_a_c(lane, qb + (size_t)(i0 + w * 16) * DH_, DH_, k0);
                bf16x16 bfr[4];
#pragma unroll
                for (int s = 0; s < 4; ++s) bfr[s] = build_b_c(lane, cb + (size_t)(j0 + s * 16) * DH_, DH_, k0);
#pragma unroll
                for (int s = 0; s < 4; ++s) acc[s] = wmma_bf16(af, bfr[s], acc[s]);
            }
#pragma unroll
            for (int s = 0; s < 4; ++s)
#pragma unroll
                for (int r = 0; r < 8; ++r) {
                    int il = w * 16 + r + 8 * half, jl = s * 16 + nl;
                    float e = __expf(acc[s][r] * SCALE_) *
                              (side ? cinv[h * 64 + jl] : rinv[h * 64 + il]);
                    ldsP[(il * 64 + jl) * 18 + h] = f2bf(e);
                }
        }
        __syncthreads();
        // Phase B: head mix as WMMA GEMM over h (K=16 live, 16 zero-padded)
        bf16x16 mixA = side ? cthfrag : thfrag;
        for (int c = w; c < 256; c += 4) {
            int e0 = c * 16;
            bf16x16 bfr = build_b(lane, [&](int k, int n) {
                return k < 16 ? ldpair(&ldsP[(e0 + n) * 18 + k]) : 0u; });
            f32x8 dz = {};
            f32x8 d = wmma_bf16(mixA, bfr, dz);
            int e = e0 + nl;
#pragma unroll
            for (int r = 0; r < 8; ++r) {
                int g = r + 8 * half;
                u16 hv = f2bf(d[r]);
                if (side == 0) ldsM[g * 4112 + e] = hv;                        // [i][j]
                else           ldsM[g * 4112 + (e & 63) * 64 + (e >> 6)] = hv; // [j][i]
            }
        }
        __syncthreads();
        // Phase C: per-head 64x64x64 WMMA GEMM against V tile, atomically accumulate
        float* outp = side ? ctx_acc : out_acc;
        const u16* vsrc = side ? vv : cv;
        int r0 = side ? j0 : i0;   // output row base
        int rS = side ? i0 : j0;   // V source row base (reduction axis)
        for (int g = 0; g < H_; ++g) {
            const u16* vb = vsrc + ((size_t)(b * H_ + g)) * NSEQ * DH_ + (size_t)rS * DH_;
#pragma unroll
            for (int q2 = 0; q2 < 4; ++q2) {       // 64x64 tile: b128 loads, transposed u16 stores
                int c2 = t + 128 * q2;
                int rr = c2 >> 3, db = (c2 & 7) * 8;
                u32x4 v4 = *(const u32x4*)(vb + (size_t)rr * DH_ + db);
                const u16* e = (const u16*)&v4;
#pragma unroll
                for (int k = 0; k < 8; ++k) Vt[(db + k) * 72 + rr] = e[k];
            }
            __syncthreads();
            f32x8 acc[4] = {};
#pragma unroll
            for (int k0 = 0; k0 < 64; k0 += 32) {
                bf16x16 af = build_a_c(lane, ldsM + g * 4112 + (w * 16) * 64, 64, k0);
                bf16x16 bfr[4];
#pragma unroll
                for (int s = 0; s < 4; ++s) bfr[s] = build_b_c(lane, Vt + (s * 16) * 72, 72, k0);
#pragma unroll
                for (int s = 0; s < 4; ++s) acc[s] = wmma_bf16(af, bfr[s], acc[s]);
            }
#pragma unroll
            for (int s = 0; s < 4; ++s)
#pragma unroll
                for (int r = 0; r < 8; ++r) {
                    int orow = r0 + w * 16 + r + 8 * half, d = s * 16 + nl;
                    atomicAdd(outp + ((size_t)(b * H_ + g) * NSEQ + orow) * DH_ + d, acc[s][r]);
                }
            __syncthreads();
        }
    }
}

// ---------------- merge heads [b,h,n,d] f32 -> [b,n,h*d] bf16 ----------------
__global__ void __launch_bounds__(256) k_merge(const float* __restrict__ acc, u16* __restrict__ out, int n) {
    int base = (blockIdx.x * 256 + threadIdx.x) * 4;
    if (base >= n) return;
    int c = base & 1023, nn = (base >> 10) & 2047, bb = base >> 21;
    int h = c >> 6, d = c & 63;
    f32x4 v = *(const f32x4*)(acc + (((size_t)(bb * H_ + h)) * NSEQ + nn) * DH_ + d);
    u32x2 o = {f2bf2(v[0], v[1]), f2bf2(v[2], v[3])};
    *(u32x2*)(out + base) = o;
}

extern "C" void kernel_launch(void* const* d_in, const int* in_sizes, int n_in,
                              void* d_out, int out_size, void* d_ws, size_t ws_size,
                              hipStream_t stream) {
    const float* x      = (const float*)d_in[0];
    const float* ctx    = (const float*)d_in[1];
    const float* ln_g   = (const float*)d_in[2];
    const float* ln_b   = (const float*)d_in[3];
    const float* cln_g  = (const float*)d_in[4];
    const float* cln_b  = (const float*)d_in[5];
    const float* W_qk   = (const float*)d_in[6];
    const float* W_cqk  = (const float*)d_in[7];
    const float* W_v    = (const float*)d_in[8];
    const float* W_cv   = (const float*)d_in[9];
    const float* W_out  = (const float*)d_in[10];
    const float* b_out  = (const float*)d_in[11];
    const float* W_cout = (const float*)d_in[12];
    const float* b_cout = (const float*)d_in[13];
    const float* th_w   = (const float*)d_in[14];
    const float* cth_w  = (const float*)d_in[15];

    char* wsp = (char*)d_ws;
    size_t off = 0;
    auto alloc = [&](size_t bytes) { size_t o = off; off = (off + bytes + 255) & ~(size_t)255; return o; };
    const size_t actE = (size_t)B_ * NSEQ * DIM_;      // 4M elems
    const size_t wE   = (size_t)DIM_ * INNER_;         // 1M elems
    const size_t hE   = (size_t)B_ * H_ * NSEQ * DH_;  // 4M elems

    u16* xn   = (u16*)(wsp + alloc(actE * 2));
    u16* cn   = (u16*)(wsp + alloc(actE * 2));
    u16* bWqk = (u16*)(wsp + alloc(wE * 2));
    u16* bWcqk= (u16*)(wsp + alloc(wE * 2));
    u16* bWv  = (u16*)(wsp + alloc(wE * 2));
    u16* bWcv = (u16*)(wsp + alloc(wE * 2));
    u16* bWo  = (u16*)(wsp + alloc(wE * 2));
    u16* bWco = (u16*)(wsp + alloc(wE * 2));
    u16* qkh  = (u16*)(wsp + alloc(hE * 2));
    u16* cqkh = (u16*)(wsp + alloc(hE * 2));
    u16* vh   = (u16*)(wsp + alloc(hE * 2));
    u16* cvh  = (u16*)(wsp + alloc(hE * 2));
    float* rowsum  = (float*)(wsp + alloc((size_t)B_ * H_ * NSEQ * 4));
    float* colsum  = (float*)(wsp + alloc((size_t)B_ * H_ * NSEQ * 4));
    float* out_acc = (float*)(wsp + alloc(hE * 4));
    float* ctx_acc = (float*)(wsp + alloc(hE * 4));
    u16* mergedA   = (u16*)(wsp + alloc(actE * 2));
    u16* mergedC   = (u16*)(wsp + alloc(actE * 2));

    // init accumulators
    k_zero4<<<(B_ * H_ * NSEQ / 4 + 255) / 256, 256, 0, stream>>>((f32x4*)colsum, B_ * H_ * NSEQ / 4);
    k_zero4<<<(int)((hE / 4 + 255) / 256), 256, 0, stream>>>((f32x4*)out_acc, (int)(hE / 4));
    k_zero4<<<(int)((hE / 4 + 255) / 256), 256, 0, stream>>>((f32x4*)ctx_acc, (int)(hE / 4));

    // layernorm -> bf16
    k_layernorm<<<B_ * NSEQ, 256, 0, stream>>>(x, ln_g, ln_b, xn);
    k_layernorm<<<B_ * NSEQ, 256, 0, stream>>>(ctx, cln_g, cln_b, cn);

    // weights -> bf16
    int wg = (int)((wE / 4 + 255) / 256);
    k_f32bf<<<wg, 256, 0, stream>>>(W_qk, bWqk, (int)wE);
    k_f32bf<<<wg, 256, 0, stream>>>(W_cqk, bWcqk, (int)wE);
    k_f32bf<<<wg, 256, 0, stream>>>(W_v, bWv, (int)wE);
    k_f32bf<<<wg, 256, 0, stream>>>(W_cv, bWcv, (int)wE);
    k_f32bf<<<wg, 256, 0, stream>>>(W_out, bWo, (int)wE);
    k_f32bf<<<wg, 256, 0, stream>>>(W_cout, bWco, (int)wE);

    // projections (bf16 WMMA, head-split output)
    dim3 gg(B_ * NSEQ / 64, INNER_ / 64);
    k_gemm<<<gg, 128, 0, stream>>>(xn, bWqk, DIM_, INNER_, 0, nullptr, nullptr, qkh);
    k_gemm<<<gg, 128, 0, stream>>>(cn, bWcqk, DIM_, INNER_, 0, nullptr, nullptr, cqkh);
    k_gemm<<<gg, 128, 0, stream>>>(xn, bWv, DIM_, INNER_, 0, nullptr, nullptr, vh);
    k_gemm<<<gg, 128, 0, stream>>>(cn, bWcv, DIM_, INNER_, 0, nullptr, nullptr, cvh);

    // attention pass 1: softmax denominators both axes
    k_attn_sums<<<dim3(B_ * H_, NSEQ / 64), 128, 0, stream>>>(qkh, cqkh, rowsum, colsum);

    // attention pass 2: normalize + talking-heads mix + P@V / Pc^T@V (big-LDS fused tile kernel)
    size_t smem = (size_t)(4096 * 18 + 16 * 4112 + 64 * 72) * 2 + 2048 * 4; // 296448 B
    k_attn_apply<<<dim3((NSEQ / 64) * (NSEQ / 64), B_), 128, smem, stream>>>(
        qkh, cqkh, vh, cvh, rowsum, colsum, th_w, cth_w, out_acc, ctx_acc);

    // merge heads -> bf16
    k_merge<<<(int)((actE / 4 + 255) / 256), 256, 0, stream>>>(out_acc, mergedA, (int)actE);
    k_merge<<<(int)((actE / 4 + 255) / 256), 256, 0, stream>>>(ctx_acc, mergedC, (int)actE);

    // output projections (f32 out + bias) straight into d_out: [out | ctx_out]
    float* outF = (float*)d_out;
    k_gemm<<<gg, 128, 0, stream>>>(mergedA, bWo, INNER_, DIM_, 1, b_out, outF, nullptr);
    k_gemm<<<gg, 128, 0, stream>>>(mergedC, bWco, INNER_, DIM_, 1, b_cout, outF + actE, nullptr);
    (void)in_sizes; (void)n_in; (void)out_size; (void)ws_size;
}
